// PCC_87883620811519
// MI455X (gfx1250) — matrix operand order repaired and checked by reading, not compiled, not run on hardware
//
#include <hip/hip_runtime.h>
#include <hip/hip_bf16.h>
#include <stdint.h>

// CDNA5 / gfx1250: wave32, WMMA 16x16x32 f16 -> f32 accumulate, TDM staging.
typedef __attribute__((ext_vector_type(16))) _Float16 v16h;
typedef __attribute__((ext_vector_type(8)))  float    v8f;
typedef __attribute__((ext_vector_type(4)))  unsigned int u32x4;
typedef __attribute__((ext_vector_type(8)))  int       i32x8;
typedef __attribute__((ext_vector_type(4)))  int       i32x4;

#define RSQ_1PEPS 0.99999500003749968750f  // 1/sqrt(1 + 1e-5)  (BatchNorm eval, var=1)

#if __has_builtin(__builtin_amdgcn_tensor_load_to_lds) && __has_builtin(__builtin_amdgcn_s_wait_tensorcnt)
#define TDM_OK 1
#endif

// ---------------------------------------------------------------------------
// f32 -> f16 convert (weights, once per call)
// ---------------------------------------------------------------------------
__global__ void f32_to_f16_kernel(const float* __restrict__ in, _Float16* __restrict__ out,
                                  long long nElem) {
  long long e = (long long)blockIdx.x * blockDim.x + threadIdx.x;
  if (e < nElem) out[e] = (_Float16)in[e];
}

// ---------------------------------------------------------------------------
// KNN: for each (b,i) find 32 nearest (top_k on d = -|xi-xj|^2, self incl.)
// ---------------------------------------------------------------------------
__global__ void knn_kernel(const float* __restrict__ pts, int* __restrict__ idx, int n) {
  __shared__ float sx[2048], sy[2048], sz[2048];
  __shared__ float sd[64][32];
  __shared__ int   si[64][32];
  const int b = blockIdx.y;
  const float* xb = pts + (size_t)b * 3 * n;
  for (int e = threadIdx.x; e < n; e += blockDim.x) {
    sx[e] = xb[e]; sy[e] = xb[n + e]; sz[e] = xb[2 * n + e];
  }
  __syncthreads();
  const int i = blockIdx.x * 64 + threadIdx.x;
  if (i >= n) return;
  const float pix = sx[i], piy = sy[i], piz = sz[i];
  float* dl = sd[threadIdx.x];
  int*   il = si[threadIdx.x];
  for (int q = 0; q < 32; ++q) { dl[q] = -3.4e38f; il[q] = 0; }
  for (int j = 0; j < n; ++j) {
    float dx = sx[j] - pix, dy = sy[j] - piy, dz = sz[j] - piz;
    float d = -(dx * dx + dy * dy + dz * dz);
    if (d > dl[31]) {
      int pos = 31;
      while (pos > 0 && d > dl[pos - 1]) { dl[pos] = dl[pos - 1]; il[pos] = il[pos - 1]; --pos; }
      dl[pos] = d; il[pos] = j;
    }
  }
  int* op = idx + ((size_t)b * n + i) * 32;
  for (int q = 0; q < 32; ++q) op[q] = il[q];
}

// ---------------------------------------------------------------------------
// Fused PCE: gather 6-d edge feats, conv1(6->64)+BN+relu (scalar),
// conv2 via WMMA (128x64 x k=32), BN+relu, max over k. One wave per point.
// ---------------------------------------------------------------------------
__global__ void pce_kernel(const float* __restrict__ pts, const int* __restrict__ idx,
                           const float* __restrict__ w1, const float* __restrict__ g1,
                           const float* __restrict__ b1, const _Float16* __restrict__ w2,
                           const float* __restrict__ g2, const float* __restrict__ b2,
                           _Float16* __restrict__ out16, int Bn, int n) {
  __shared__ _Float16 sW2[128 * 64];
  __shared__ float    sW1[64 * 6];
  __shared__ _Float16 sH1[4][64 * 32];
  const int tid = threadIdx.x;
  const int wave = tid >> 5, lane = tid & 31;
  for (int e = tid; e < 128 * 64; e += 128) sW2[e] = w2[e];
  for (int e = tid; e < 64 * 6; e += 128) sW1[e] = w1[e];
  __syncthreads();

  const int p = blockIdx.x * 4 + wave;
  const int b = p / n, i = p - b * n;
  float f0, f1, f2, f3, f4, f5;
  {
    const int j = idx[(size_t)p * 32 + lane];
    const float* xb = pts + (size_t)b * 3 * n;
    const float pix = xb[i], piy = xb[n + i], piz = xb[2 * n + i];
    const float pjx = xb[j], pjy = xb[n + j], pjz = xb[2 * n + j];
    f0 = pjx - pix; f1 = pjy - piy; f2 = pjz - piz;
    f3 = pjx; f4 = pjy; f5 = pjz;
  }
  _Float16* h1 = sH1[wave];
  for (int c = 0; c < 64; ++c) {
    const float* wr = &sW1[c * 6];
    float s = wr[0]*f0 + wr[1]*f1 + wr[2]*f2 + wr[3]*f3 + wr[4]*f4 + wr[5]*f5;
    s = fmaxf(s * (g1[c] * RSQ_1PEPS) + b1[c], 0.0f);
    h1[c * 32 + lane] = (_Float16)s;
  }
  __syncthreads();

  const int r = lane & 15, hs = lane >> 4;
  for (int mt = 0; mt < 8; ++mt) {
    v8f acc0 = {}, acc1 = {};
    #pragma unroll
    for (int k0 = 0; k0 < 64; k0 += 32) {
      v16h av, bv0, bv1;
      const _Float16* Ap = sW2 + (mt * 16 + r) * 64 + k0;
      #pragma unroll
      for (int v = 0; v < 8; ++v) {
        const int kb = ((v & 4) << 2) + hs * 8 + ((v & 3) << 1);
        av[2 * v]     = Ap[kb];
        av[2 * v + 1] = Ap[kb + 1];
      }
      const int ks = hs * 8 + k0;
      #pragma unroll
      for (int v = 0; v < 8; ++v) {
        const int kr = ks + ((v & 4) << 2) + ((v & 3) << 1);
        bv0[2 * v]     = h1[(kr) * 32 + r];
        bv0[2 * v + 1] = h1[(kr + 1) * 32 + r];
        bv1[2 * v]     = h1[(kr) * 32 + 16 + r];
        bv1[2 * v + 1] = h1[(kr + 1) * 32 + 16 + r];
      }
      acc0 = __builtin_amdgcn_wmma_f32_16x16x32_f16(false, av, false, bv0, (short)0, acc0, false, false);
      acc1 = __builtin_amdgcn_wmma_f32_16x16x32_f16(false, av, false, bv1, (short)0, acc1, false, false);
    }
    #pragma unroll
    for (int v = 0; v < 8; ++v) {
      const int m = mt * 16 + v + hs * 8;
      const float sc = g2[m] * RSQ_1PEPS, sh = b2[m];
      float mm = fmaxf(fmaxf(acc0[v] * sc + sh, 0.0f), fmaxf(acc1[v] * sc + sh, 0.0f));
      for (int o = 1; o < 16; o <<= 1) mm = fmaxf(mm, __shfl_xor(mm, o, 32));
      if (r == 0) out16[(size_t)m * Bn + p] = (_Float16)mm;
    }
  }
}

// ---------------------------------------------------------------------------
// Tiled WMMA GEMM: C(MxN) = epilogue( A(MxK) * B(KxN) ).
// 256 threads = 8 waves; block tile 128x64 (wave grid 2x4, 64x16 per wave ->
// 4 WMMAs per wave per 32-wide k-step). A/B panels staged in LDS:
//   - A: vector b128 copies when acs==1, else scalar (handles transposed
//     attention-score view); zero-padded when M not a multiple of 128.
//   - B: Tensor Data Mover (tensor_load_to_lds, 2D D#, TENSORcnt) when
//     bcs==1, else vector/scalar copies.
// Epilogue: (+bias) -> BN(gamma/sqrt(1+eps), beta) -> act -> (+resid),
// dual fp32/fp16 outputs. M any, N % 64 == 0, K % 32 == 0.
// ---------------------------------------------------------------------------
#define GEMM_BM 128
#define GEMM_BN 64
#define GEMM_BK 32

__global__ __launch_bounds__(256) void wmma_gemm_kernel(
    const _Float16* __restrict__ A, long long ars, long long acs,
    const _Float16* __restrict__ B, long long brs, long long bcs,
    float* __restrict__ C32, _Float16* __restrict__ C16, long long ldc,
    const float* __restrict__ bias,
    const float* __restrict__ gamma, const float* __restrict__ beta,
    const float* __restrict__ resid, long long ldr,
    int M, int N, int K, int act, int useTdm) {
  __shared__ __align__(16) _Float16 sA[GEMM_BM * GEMM_BK];  // 8 KB, K contiguous
  __shared__ __align__(16) _Float16 sB[GEMM_BK * GEMM_BN];  // 4 KB, N contiguous
  const int tid = threadIdx.x;
  const int wave = tid >> 5, lane = tid & 31;
  const int r = lane & 15, hs = lane >> 4;
  const int waveM = wave >> 2, waveN = wave & 3;
  const int tM = blockIdx.x * GEMM_BM;
  const int tN = blockIdx.y * GEMM_BN;

  v8f acc[4] = {};
  for (int k0 = 0; k0 < K; k0 += GEMM_BK) {
    __syncthreads();
    // ---------------- stage A panel ----------------
    if (acs == 1) {
      #pragma unroll
      for (int t = 0; t < 2; ++t) {
        const int e = tid + t * 256;               // 512 x uint4
        const int row = e >> 2, q = e & 3;
        uint4 val = make_uint4(0u, 0u, 0u, 0u);
        if (tM + row < M)
          val = *(const uint4*)(A + (long long)(tM + row) * ars + k0 + q * 8);
        *(uint4*)(sA + row * GEMM_BK + q * 8) = val;
      }
    } else {
      #pragma unroll
      for (int t = 0; t < 16; ++t) {
        const int e = tid + t * 256;               // 4096 halfs
        const int row = e >> 5, kk = e & 31;
        _Float16 val = (_Float16)0.0f;
        if (tM + row < M)
          val = A[(long long)(tM + row) * ars + (long long)(k0 + kk) * acs];
        sA[row * GEMM_BK + kk] = val;
      }
    }
    // ---------------- stage B panel ----------------
#ifdef TDM_OK
    if (useTdm) {
      if (wave == 0) {
        const _Float16* gsrc = B + (long long)k0 * brs + tN;
        const unsigned long long ga = (unsigned long long)(uintptr_t)gsrc;
        const unsigned ldsAddr = (unsigned)(uintptr_t)(&sB[0]);  // flat addr[31:0] = LDS offset
        // D# group 0: count=1 | lds_addr | global_addr | type=2
        u32x4 g0;
        g0[0] = 1u;                                   // count = 1 valid descriptor
        g0[1] = ldsAddr;                              // lds_addr (bytes)
        g0[2] = (unsigned)(ga & 0xFFFFFFFFull);       // global_addr[31:0]
        g0[3] = (unsigned)((ga >> 32) & 0x01FFFFFFull) | (2u << 30);  // addr[56:32] | type=2
        // D# group 1: data_size=2B, tensor (N rows-len, K rows), tile 64x32
        const unsigned long long td0 = (unsigned long long)(unsigned)N;   // tensor_dim0
        const unsigned td1 = (unsigned)K;                                 // tensor_dim1
        const unsigned long long st0 = (unsigned long long)brs;           // dim0 stride
        i32x8 g1;
        g1[0] = (int)(1u << 16);                                   // data_size = 1 (2 bytes)
        g1[1] = (int)(((unsigned)td0 & 0xFFFFu) << 16);            // tensor_dim0[15:0]
        g1[2] = (int)(((unsigned)td0 >> 16) | ((td1 & 0xFFFFu) << 16));
        g1[3] = (int)((td1 >> 16) | ((unsigned)GEMM_BN << 16));    // tile_dim0 = 64
        g1[4] = (int)((unsigned)GEMM_BK);                          // tile_dim1 = 32, tile_dim2 = 0
        g1[5] = (int)(unsigned)(st0 & 0xFFFFFFFFull);              // dim0_stride lo32
        g1[6] = (int)(unsigned)((st0 >> 32) & 0xFFFFull);          // dim0_stride hi16
        g1[7] = 0;
        i32x4 gz = {0, 0, 0, 0};
#if defined(__clang_major__) && (__clang_major__ >= 23)
        i32x8 gz8 = {0, 0, 0, 0, 0, 0, 0, 0};
        __builtin_amdgcn_tensor_load_to_lds(g0, g1, gz, gz, gz8, 0);
#else
        __builtin_amdgcn_tensor_load_to_lds(g0, g1, gz, gz, 0);
#endif
        __builtin_amdgcn_s_wait_tensorcnt(0);
      }
    } else
#endif
    if (bcs == 1) {
      const int krow = tid >> 3, q = tid & 7;      // 32 rows x 8 uint4
      *(uint4*)(sB + krow * GEMM_BN + q * 8) =
          *(const uint4*)(B + (long long)(k0 + krow) * brs + tN + q * 8);
    } else {
      #pragma unroll
      for (int t = 0; t < 8; ++t) {
        const int e = tid + t * 256;               // 2048 halfs
        const int krow = e >> 6, col = e & 63;
        sB[krow * GEMM_BN + col] =
            B[(long long)(k0 + krow) * brs + (long long)(tN + col) * bcs];
      }
    }
    __syncthreads();
    // ---------------- fragments + 4 WMMAs ----------------
    v16h bv;
    const int bcol = waveN * 16 + r;
    const int ks = hs * 8;
    #pragma unroll
    for (int v = 0; v < 8; ++v) {
      const int kr = ks + ((v & 4) << 2) + ((v & 3) << 1);
      bv[2 * v]     = sB[(kr) * GEMM_BN + bcol];
      bv[2 * v + 1] = sB[(kr + 1) * GEMM_BN + bcol];
    }
    #pragma unroll
    for (int sub = 0; sub < 4; ++sub) {
      v16h av;
      const _Float16* Ap = sA + (waveM * 64 + sub * 16 + r) * GEMM_BK;
      #pragma unroll
      for (int v = 0; v < 8; ++v) {
        const int kb = ((v & 4) << 2) + hs * 8 + ((v & 3) << 1);
        av[2 * v]     = Ap[kb];
        av[2 * v + 1] = Ap[kb + 1];
      }
      acc[sub] = __builtin_amdgcn_wmma_f32_16x16x32_f16(false, av, false, bv, (short)0,
                                                        acc[sub], false, false);
    }
  }
  // ---------------- epilogue ----------------
  const int col = tN + waveN * 16 + r;
  #pragma unroll
  for (int sub = 0; sub < 4; ++sub) {
    #pragma unroll
    for (int v = 0; v < 8; ++v) {
      const int m = tM + waveM * 64 + sub * 16 + v + hs * 8;
      if (m < M) {
        float val = acc[sub][v];
        if (bias)  val += bias[m];
        if (gamma) val = val * (gamma[m] * RSQ_1PEPS) + (beta ? beta[m] : 0.0f);
        if (act == 1)      val = fmaxf(val, 0.0f);
        else if (act == 2) val = (val >= 0.0f) ? val : 0.2f * val;
        if (resid) val += resid[(long long)m * ldr + col];
        if (C32) C32[(long long)m * ldc + col] = val;
        if (C16) C16[(long long)m * ldc + col] = (_Float16)val;
      }
    }
  }
}

// ---------------------------------------------------------------------------
// Row softmax (in place, fp32). One block (256 threads) per row of length n.
// ---------------------------------------------------------------------------
__global__ void softmax_row_kernel(float* __restrict__ att, int n) {
  __shared__ float red[256];
  float* row = att + (size_t)blockIdx.x * n;
  float mx = -3.4e38f;
  for (int j = threadIdx.x; j < n; j += 256) mx = fmaxf(mx, row[j]);
  red[threadIdx.x] = mx; __syncthreads();
  for (int s = 128; s > 0; s >>= 1) {
    if (threadIdx.x < s) red[threadIdx.x] = fmaxf(red[threadIdx.x], red[threadIdx.x + s]);
    __syncthreads();
  }
  mx = red[0]; __syncthreads();
  float sum = 0.0f;
  for (int j = threadIdx.x; j < n; j += 256) { float e = __expf(row[j] - mx); row[j] = e; sum += e; }
  red[threadIdx.x] = sum; __syncthreads();
  for (int s = 128; s > 0; s >>= 1) {
    if (threadIdx.x < s) red[threadIdx.x] += red[threadIdx.x + s];
    __syncthreads();
  }
  const float inv = 1.0f / red[0];
  for (int j = threadIdx.x; j < n; j += 256) row[j] *= inv;
}

// Column sums over the n x n attention (L1 renorm over queries).
__global__ void colsum_kernel(const float* __restrict__ att, float* __restrict__ cs, int n) {
  const int m = blockIdx.x * blockDim.x + threadIdx.x;
  if (m >= n) return;
  float s = 0.0f;
  for (int i = 0; i < n; ++i) s += att[(size_t)i * n + m];
  cs[m] = s;
}

// att16 = att / (1e-9 + colsum[col])
__global__ void attscale_kernel(const float* __restrict__ att, const float* __restrict__ cs,
                                _Float16* __restrict__ att16, int n) {
  const long long e = (long long)blockIdx.x * blockDim.x + threadIdx.x;
  const long long total = (long long)n * n;
  if (e >= total) return;
  const int m = (int)(e % n);
  att16[e] = (_Float16)(att[e] / (1e-9f + cs[m]));
}

// diff16 = (fp16)(a - b)
__global__ void sub_f16_kernel(const float* __restrict__ a, const float* __restrict__ b,
                               _Float16* __restrict__ o, long long total) {
  const long long e = (long long)blockIdx.x * blockDim.x + threadIdx.x;
  if (e < total) o[e] = (_Float16)(a[e] - b[e]);
}

// gate stage 2: gw[j] = sigmoid( dot(w2[256], h[:,j]) + b2 )
__global__ void gate2_kernel(const _Float16* __restrict__ h, const float* __restrict__ w2,
                             const float* __restrict__ b2, float* __restrict__ gw, int N) {
  const int j = blockIdx.x * blockDim.x + threadIdx.x;
  if (j >= N) return;
  float s = b2[0];
  for (int c = 0; c < 256; ++c) s += (float)h[(size_t)c * N + j] * w2[c];
  gw[j] = 1.0f / (1.0f + __expf(-s));
}

// pooled[b][m] = max_i f[m][b*n+i] * gw[b*n+i]
__global__ void gatedmax_kernel(const float* __restrict__ f, const float* __restrict__ gw,
                                float* __restrict__ out, int n, int N, int ostride) {
  const int t = blockIdx.x * blockDim.x + threadIdx.x;
  const int Bsz = N / n;
  if (t >= Bsz * 512) return;
  const int b = t >> 9, m = t & 511;
  const float* row = f + (size_t)m * N + (size_t)b * n;
  const float* g = gw + (size_t)b * n;
  float mx = -3.4e38f;
  for (int i = 0; i < n; ++i) mx = fmaxf(mx, row[i] * g[i]);
  out[(size_t)b * ostride + m] = mx;
}

// Small dense layer (batch = 8)
__global__ void dense_kernel(const float* __restrict__ in, const float* __restrict__ W,
                             const float* __restrict__ bias, const float* __restrict__ g,
                             const float* __restrict__ sh, float* __restrict__ out,
                             int Bsz, int O, int K, int act) {
  const int t = blockIdx.x * blockDim.x + threadIdx.x;
  if (t >= Bsz * O) return;
  const int b = t / O, o = t - b * O;
  const float* wr = W + (size_t)o * K;
  const float* xr = in + (size_t)b * K;
  float s = 0.0f;
  for (int c = 0; c < K; ++c) s += wr[c] * xr[c];
  if (bias) s += bias[o];
  if (g) s = s * (g[o] * RSQ_1PEPS) + (sh ? sh[o] : 0.0f);
  if (act == 2) s = (s >= 0.0f) ? s : 0.2f * s;
  out[t] = s;
}

// ---------------------------------------------------------------------------
// Host orchestration
// ---------------------------------------------------------------------------
extern "C" void kernel_launch(void* const* d_in, const int* in_sizes, int n_in,
                              void* d_out, int out_size, void* d_ws, size_t ws_size,
                              hipStream_t stream) {
  if (n_in < 77) return;
  const int B = 8, NMAX = 2048, BNMAX = B * NMAX;

  auto F = [&](int i) { return (const float*)d_in[i]; };
  const float* x_in   = F(0);
  const float* vx1_in = F(1);
  const float* vx2_in = F(2);
  // d_in[3] = k (32; kernels are structured around wave32 == k)

  // params tree-flattened with sorted keys (jax.tree_util order):
  enum {
    I_bn1_b = 4, I_bn1_g, I_bn2_b, I_bn2_g, I_fuse_b, I_fuse_g, I_fuse_w,
    I_hf_b, I_hf_g, I_hf_w, I_lin1_w, I_lin2_b, I_lin2_w, I_lin3_b, I_lin3_w,
    I_pce_b1, I_pce_b2, I_pce_g1, I_pce_g2, I_pce_w1, I_pce_w2,
    I_pt_b1, I_pt_b2, I_pt_g1, I_pt_g2, I_pt_w1, I_pt_w2,
    I_sa1 = 31,  // per sa (7): nb, ng, qk, tb, tw, vb, vw
    I_wl1 = 59   // per wl (6): b, b1, b2, g, w1, w2
  };

  // ---- workspace layout ----
  char* ws = (char*)d_ws;
  size_t off = 0;
  auto alloc = [&](size_t bytes) -> char* {
    off = (off + 255) & ~(size_t)255;
    char* p = ws + off;
    off += bytes;
    return p;
  };
  int*      idxbuf    = (int*)alloc((size_t)BNMAX * 32 * 4);
  _Float16* pce_w2_16 = (_Float16*)alloc(128 * 64 * 2);
  _Float16* pt_w1_16  = (_Float16*)alloc(128 * 128 * 2);
  _Float16* pt_w2_16  = (_Float16*)alloc(128 * 128 * 2);
  _Float16 *qk16[4], *vw16[4], *tw16[4];
  for (int l = 0; l < 4; ++l) {
    qk16[l] = (_Float16*)alloc(32 * 128 * 2);
    vw16[l] = (_Float16*)alloc(128 * 128 * 2);
    tw16[l] = (_Float16*)alloc(128 * 128 * 2);
  }
  _Float16* fuse_w16 = (_Float16*)alloc((size_t)512 * 640 * 2);
  _Float16* wlw1_16[3];
  for (int c = 0; c < 3; ++c) wlw1_16[c] = (_Float16*)alloc((size_t)256 * 512 * 2);

  _Float16* xcat16 = (_Float16*)alloc((size_t)640 * BNMAX * 2);  // [pce, x1..x4]
  _Float16* h16    = (_Float16*)alloc((size_t)128 * BNMAX * 2);
  _Float16* x16    = (_Float16*)alloc((size_t)128 * BNMAX * 2);
  float*    xa32   = (float*)alloc((size_t)128 * BNMAX * 4);
  float*    xb32   = (float*)alloc((size_t)128 * BNMAX * 4);
  _Float16* xqk16  = (_Float16*)alloc((size_t)32 * BNMAX * 2);
  _Float16* xv16   = (_Float16*)alloc((size_t)128 * BNMAX * 2);
  float*    att32  = (float*)alloc((size_t)NMAX * NMAX * 4);
  _Float16* att16  = (_Float16*)alloc((size_t)NMAX * NMAX * 2);
  float*    colsum = (float*)alloc((size_t)NMAX * 4);
  float*    xr32   = (float*)alloc((size_t)128 * BNMAX * 4);
  _Float16* diff16 = (_Float16*)alloc((size_t)128 * BNMAX * 2);
  float*    f032   = (float*)alloc((size_t)512 * BNMAX * 4);
  _Float16* f016   = (_Float16*)alloc((size_t)512 * BNMAX * 2);
  _Float16* gh16   = (_Float16*)alloc((size_t)256 * BNMAX * 2);
  float*    gatew  = (float*)alloc((size_t)BNMAX * 4);
  float*    hbuf   = (float*)alloc((size_t)8 * 1536 * 4);
  float*    t1     = (float*)alloc((size_t)8 * 1024 * 4);
  float*    t2     = (float*)alloc((size_t)8 * 512 * 4);
  float*    t3     = (float*)alloc((size_t)8 * 256 * 4);
  if (off > ws_size) return;

  auto cvt = [&](const float* src, _Float16* dst, long long nE) {
    f32_to_f16_kernel<<<(int)((nE + 255) / 256), 256, 0, stream>>>(src, dst, nE);
  };
  cvt(F(I_pce_w2), pce_w2_16, 128 * 64);
  cvt(F(I_pt_w1), pt_w1_16, 128 * 128);
  cvt(F(I_pt_w2), pt_w2_16, 128 * 128);
  for (int l = 0; l < 4; ++l) {
    const int sb = I_sa1 + 7 * l;
    cvt(F(sb + 2), qk16[l], 32 * 128);
    cvt(F(sb + 4), tw16[l], 128 * 128);
    cvt(F(sb + 6), vw16[l], 128 * 128);
  }
  cvt(F(I_fuse_w), fuse_w16, (long long)512 * 640);
  for (int c = 0; c < 3; ++c) cvt(F(I_wl1 + 6 * c + 4), wlw1_16[c], (long long)256 * 512);

  auto gemm = [&](const _Float16* A, long long ars, long long acs,
                  const _Float16* Bm, long long brs, long long bcs,
                  float* C32, _Float16* C16, long long ldc,
                  const float* bias, const float* g, const float* sh,
                  const float* resid, long long ldr,
                  int M, int N, int K, int act) {
    dim3 grid((M + GEMM_BM - 1) / GEMM_BM, N / GEMM_BN, 1);
    const int useTdm = (bcs == 1) ? 1 : 0;  // TDM stages contiguous-row B panels
    wmma_gemm_kernel<<<grid, dim3(256), 0, stream>>>(A, ars, acs, Bm, brs, bcs,
                                                     C32, C16, ldc, bias, g, sh,
                                                     resid, ldr, M, N, K, act, useTdm);
  };

  auto run_cloud = [&](const float* pts, int n, int cloud) {
    const int N = B * n;
    knn_kernel<<<dim3(n / 64, B), 64, 0, stream>>>(pts, idxbuf, n);
    pce_kernel<<<N / 4, 128, 0, stream>>>(pts, idxbuf, F(I_pce_w1), F(I_pce_g1), F(I_pce_b1),
                                          pce_w2_16, F(I_pce_g2), F(I_pce_b2), xcat16, N, n);
    // point-transformer stem: two 128x128 1x1 convs with BN+relu
    gemm(pt_w1_16, 128, 1, xcat16, N, 1, nullptr, h16, N,
         nullptr, F(I_pt_g1), F(I_pt_b1), nullptr, 0, 128, N, 128, 1);
    gemm(pt_w2_16, 128, 1, h16, N, 1, xa32, x16, N,
         nullptr, F(I_pt_g2), F(I_pt_b2), nullptr, 0, 128, N, 128, 1);

    for (int l = 0; l < 4; ++l) {
      const int sb = I_sa1 + 7 * l;
      const float* s_nb = F(sb + 0);
      const float* s_ng = F(sb + 1);
      const float* s_tb = F(sb + 3);
      const float* s_vb = F(sb + 5);
      float* xin32  = (l & 1) ? xb32 : xa32;
      float* xout32 = (l & 1) ? xa32 : xb32;
      _Float16* xin16 = (l == 0) ? x16 : (xcat16 + (size_t)128 * l * N);

      // q/k projection (tied weights) and v projection
      gemm(qk16[l], 128, 1, xin16, N, 1, nullptr, xqk16, N,
           nullptr, nullptr, nullptr, nullptr, 0, 32, N, 128, 0);
      gemm(vw16[l], 128, 1, xin16, N, 1, nullptr, xv16, N,
           s_vb, nullptr, nullptr, nullptr, 0, 128, N, 128, 0);

      for (int b = 0; b < B; ++b) {
        // scores: att[i][j] = sum_c q[c][i] k[c][j]  (A = qk^T via strides)
        gemm(xqk16 + (size_t)b * n, 1, N, xqk16 + (size_t)b * n, N, 1,
             att32, nullptr, n, nullptr, nullptr, nullptr, nullptr, 0, n, n, 32, 0);
        softmax_row_kernel<<<n, 256, 0, stream>>>(att32, n);
        colsum_kernel<<<(n + 255) / 256, 256, 0, stream>>>(att32, colsum, n);
        attscale_kernel<<<(int)(((long long)n * n + 255) / 256), 256, 0, stream>>>(att32, colsum, att16, n);
        // xr[c][m] = sum_i xv[c][i] * att[i][m]
        gemm(xv16 + (size_t)b * n, N, 1, att16, n, 1, xr32 + (size_t)b * n, nullptr, N,
             nullptr, nullptr, nullptr, nullptr, 0, 128, n, n, 0);
      }
      sub_f16_kernel<<<(int)(((long long)128 * N + 255) / 256), 256, 0, stream>>>(
          xin32, xr32, diff16, (long long)128 * N);
      // out = x + relu(bn(tw*(x-xr)+tb)); fp32 -> ping-pong, fp16 -> concat block l+1
      gemm(tw16[l], 128, 1, diff16, N, 1, xout32, xcat16 + (size_t)128 * (l + 1) * N, N,
           s_tb, s_ng, s_nb, xin32, N, 128, N, 128, 1);
    }

    // fuse: 512x640 over concat, lrelu(bn)
    gemm(fuse_w16, 640, 1, xcat16, N, 1, f032, f016, N,
         nullptr, F(I_fuse_g), F(I_fuse_b), nullptr, 0, 512, N, 640, 2);

    // weight gate + gated max pool
    const int wb = I_wl1 + 6 * cloud;  // b, b1, b2, g, w1, w2
    gemm(wlw1_16[cloud], 512, 1, f016, N, 1, nullptr, gh16, N,
         F(wb + 1), F(wb + 3), F(wb + 0), nullptr, 0, 256, N, 512, 2);
    gate2_kernel<<<(N + 255) / 256, 256, 0, stream>>>(gh16, F(wb + 5), F(wb + 2), gatew, N);
    gatedmax_kernel<<<(B * 512 + 255) / 256, 256, 0, stream>>>(
        f032, gatew, hbuf + cloud * 512, n, N, 1536);
  };

  run_cloud(x_in, 2048, 0);
  run_cloud(vx1_in, 1024, 1);
  run_cloud(vx2_in, 512, 2);

  // classification head (batch = 8, tiny — scalar dot kernels)
  dense_kernel<<<(8 * 1024 + 255) / 256, 256, 0, stream>>>(
      hbuf, F(I_hf_w), nullptr, F(I_hf_g), F(I_hf_b), t1, 8, 1024, 1536, 2);
  dense_kernel<<<(8 * 512 + 255) / 256, 256, 0, stream>>>(
      t1, F(I_lin1_w), nullptr, F(I_bn1_g), F(I_bn1_b), t2, 8, 512, 1024, 2);
  dense_kernel<<<(8 * 256 + 255) / 256, 256, 0, stream>>>(
      t2, F(I_lin2_w), F(I_lin2_b), F(I_bn2_g), F(I_bn2_b), t3, 8, 256, 512, 2);
  dense_kernel<<<(8 * 40 + 255) / 256, 256, 0, stream>>>(
      t3, F(I_lin3_w), F(I_lin3_b), nullptr, nullptr, (float*)d_out, 8, 40, 256, 0);
}
